// SpotGuidedAggregation_74388833566806
// MI455X (gfx1250) — matrix operand order repaired and checked by reading, not compiled, not run on hardware
//
#include <hip/hip_runtime.h>
#include <hip/hip_bf16.h>

// ---------------------------------------------------------------------------
// SpotGuidedAggregation on gfx1250 (MI455X, wave32, WMMA).
//
// Pipeline:
//  1) l2-normalize imf/pcf rows -> f16            (l2norm_rows)
//  2) init packed argmax accumulators             (init_best)
//  3) fused f16 WMMA GEMM (5120x5120x128) with row/col argmax via
//     packed u64 atomicMax; 64x64 register-blocked per wave (16 WMMAs/chunk)
//  4) 25-NN over 3D points, points staged in LDS  (knn25)
//  5) image/pc local 25-way cosine softmax        (im_local_sim / pc_local_sim)
//  6) select = sim_score * patched best_score     (select_kernel)
//  7) seeding + spotting + unique/top_k reduce    (im_seed_spot / pc_seed_spot)
// ---------------------------------------------------------------------------

#define HH 40
#define WW 128
#define PP (HH * WW)     // 5120
#define NN PP            // 5120
#define CC 128
#define KK 25
#define SPOT 5
#define NEGV (-1e8f)

typedef __attribute__((ext_vector_type(16))) _Float16 v16h;
typedef __attribute__((ext_vector_type(8)))  float    v8f;

// -------- packed (score,index) key helpers: max == (max score, min index) ---
__device__ __forceinline__ unsigned int fkey(float f) {
    unsigned int b = __float_as_uint(f);
    return (b & 0x80000000u) ? ~b : (b | 0x80000000u);
}
__device__ __forceinline__ float funkey(unsigned int k) {
    unsigned int b = (k & 0x80000000u) ? (k & 0x7FFFFFFFu) : ~k;
    return __uint_as_float(b);
}
__device__ __forceinline__ unsigned long long packkey(float f, int idx) {
    return ((unsigned long long)fkey(f) << 32) |
           (unsigned long long)(~(unsigned int)idx);
}
__device__ __forceinline__ int key_idx(unsigned long long k) {
    return (int)(~(unsigned int)(k & 0xFFFFFFFFull));
}
__device__ __forceinline__ float key_score(unsigned long long k) {
    return funkey((unsigned int)(k >> 32));
}
__device__ __forceinline__ int iclamp(int v, int lo, int hi) {
    return v < lo ? lo : (v > hi ? hi : v);
}

// ---------------------------------------------------------------------------
// 1) row-wise L2 normalize (C=128, one block of 128 threads per row) -> f16
// ---------------------------------------------------------------------------
__global__ __launch_bounds__(128) void l2norm_rows(
    const float* __restrict__ in, _Float16* __restrict__ out) {
    __shared__ float red[4];
    int row = blockIdx.x;
    int t = threadIdx.x;
    float v = in[row * CC + t];
    float ss = v * v;
    #pragma unroll
    for (int m = 1; m < 32; m <<= 1) ss += __shfl_xor(ss, m, 32);
    if ((t & 31) == 0) red[t >> 5] = ss;
    __syncthreads();
    float tot = red[0] + red[1] + red[2] + red[3];
    float scale = rsqrtf(fmaxf(tot, 1e-24f));
    out[row * CC + t] = (_Float16)(v * scale);
}

// ---------------------------------------------------------------------------
// 2) init argmax accumulators (key 0 < any finite-score key)
// ---------------------------------------------------------------------------
__global__ void init_best(unsigned long long* __restrict__ rowbest,
                          unsigned long long* __restrict__ colbest) {
    int i = blockIdx.x * blockDim.x + threadIdx.x;
    if (i < PP) { rowbest[i] = 0ull; colbest[i] = 0ull; }
}

// ---------------------------------------------------------------------------
// 3) fused GEMM + row/col argmax. One wave per 64x64 block (4x4 tiles of
//    16x16). A = l2n(imf) (P x 128 f16), B = l2n(pcf) (N x 128 f16).
//    16x v_wmma_f32_16x16x32_f16 per K-chunk against 4 A + 4 B fragments:
//    4x fragment reuse -> fragment traffic matches WMMA compute roofline.
//    All lanes active everywhere (WMMA EXEC all-1s requirement).
// ---------------------------------------------------------------------------
__global__ __launch_bounds__(128) void gemm_argmax(
    const _Float16* __restrict__ A, const _Float16* __restrict__ B,
    unsigned long long* __restrict__ rowbest,
    unsigned long long* __restrict__ colbest) {
    const int BT = NN / 64;  // 80 block-tiles per dimension
    int wave = blockIdx.x * (blockDim.x >> 5) + (threadIdx.x >> 5);
    int tm = (wave / BT) * 64;
    int tn = (wave % BT) * 64;
    int l    = threadIdx.x & 31;
    int half = l >> 4;   // half-wave select
    int lm   = l & 15;

    v8f acc[4][4] = {};
    for (int k0 = 0; k0 < CC; k0 += 32) {
        union { v16h h; unsigned int u[8]; } a[4], b[4];
        #pragma unroll
        for (int t = 0; t < 4; ++t) {
            const _Float16* arow = A + (tm + t * 16 + lm) * CC + k0;
            const _Float16* brow = B + (tn + t * 16 + lm) * CC + k0;
            #pragma unroll
            for (int vg = 0; vg < 8; ++vg) {
                // ISA 7.12.2: 16-bit A 16x32 layout
                int ka = vg * 2 + (vg < 4 ? 0 : 8) + (half ? 8 : 0);
                // 16-bit B K x 16 layout: same k order as A (table order)
                int kb = vg * 2 + (vg < 4 ? 0 : 8) + (half ? 8 : 0);
                a[t].u[vg] = *(const unsigned int*)(arow + ka);
                b[t].u[vg] = *(const unsigned int*)(brow + kb);
            }
        }
        #pragma unroll
        for (int i = 0; i < 4; ++i)
            #pragma unroll
            for (int j = 0; j < 4; ++j)
                acc[i][j] = __builtin_amdgcn_wmma_f32_16x16x32_f16(
                    false, a[i].h, false, b[j].h, (short)0, acc[i][j],
                    false, false);
    }

    // ---- row argmax: fold 4 j-tiles locally, then 16-lane shuffle tree ----
    #pragma unroll
    for (int i = 0; i < 4; ++i) {
        #pragma unroll
        for (int r = 0; r < 8; ++r) {
            int gm = tm + i * 16 + r + half * 8;
            unsigned long long pk = 0ull;
            #pragma unroll
            for (int j = 0; j < 4; ++j) {
                unsigned long long q = packkey(acc[i][j][r], tn + j * 16 + lm);
                if (q > pk) pk = q;
            }
            #pragma unroll
            for (int m = 1; m < 16; m <<= 1) {  // xor<16 stays in half-wave
                unsigned long long o = __shfl_xor(pk, m, 32);
                if (o > pk) pk = o;
            }
            if (lm == 0) atomicMax(&rowbest[gm], pk);
        }
    }
    // ---- col argmax: fold i,r locally, merge half-waves via xor 16 ----
    #pragma unroll
    for (int j = 0; j < 4; ++j) {
        unsigned long long ck = 0ull;
        #pragma unroll
        for (int i = 0; i < 4; ++i)
            #pragma unroll
            for (int r = 0; r < 8; ++r) {
                unsigned long long q =
                    packkey(acc[i][j][r], tm + i * 16 + r + half * 8);
                if (q > ck) ck = q;
            }
        unsigned long long o = __shfl_xor(ck, 16, 32);
        if (o > ck) ck = o;
        if (half == 0) atomicMax(&colbest[tn + j * 16 + lm], ck);
    }
}

// ---------------------------------------------------------------------------
// 4) brute-force 25-NN over 5120 3D points; points staged in LDS (60 KB).
//    Tie-break: (d2, index) ascending -> matches top_k(-d2) semantics.
// ---------------------------------------------------------------------------
__global__ __launch_bounds__(256) void knn25(
    const float* __restrict__ pts, int* __restrict__ nb) {
    __shared__ float sp[NN * 3];
    for (int i = threadIdx.x; i < NN * 3; i += 256) sp[i] = pts[i];
    __syncthreads();
    int q = blockIdx.x * 256 + threadIdx.x;
    float qx = sp[q * 3 + 0], qy = sp[q * 3 + 1], qz = sp[q * 3 + 2];
    float bd[KK]; int bi[KK];
    #pragma unroll
    for (int j = 0; j < KK; ++j) { bd[j] = 3.4e38f; bi[j] = 0x7FFFFFFF; }
    for (int i = 0; i < NN; ++i) {
        float dx = sp[i * 3 + 0] - qx;
        float dy = sp[i * 3 + 1] - qy;
        float dz = sp[i * 3 + 2] - qz;
        float d2 = dx * dx + dy * dy + dz * dz;
        if (d2 < bd[KK - 1] || (d2 == bd[KK - 1] && i < bi[KK - 1])) {
            int j = KK - 1;
            while (j > 0 && (d2 < bd[j - 1] || (d2 == bd[j - 1] && i < bi[j - 1]))) {
                bd[j] = bd[j - 1]; bi[j] = bi[j - 1]; --j;
            }
            bd[j] = d2; bi[j] = i;
        }
    }
    for (int j = 0; j < KK; ++j) nb[q * KK + j] = bi[j];
}

// ---------------------------------------------------------------------------
// 5a) image local cosine softmax: one wave per pixel, lane k -> neighbor k.
// ---------------------------------------------------------------------------
__global__ __launch_bounds__(32) void im_local_sim(
    const _Float16* __restrict__ F, float* __restrict__ simsc) {
    int p = blockIdx.x;
    int k = threadIdx.x;
    int h = p / WW, w = p % WW;
    float d = -3.4e38f;
    if (k < KK) {
        int hn = iclamp(h + k / 5 - 2, 0, HH - 1);
        int wn = iclamp(w + k % 5 - 2, 0, WW - 1);
        const _Float16* a = F + p * CC;
        const _Float16* b = F + (hn * WW + wn) * CC;
        float acc = 0.f;
        for (int c2 = 0; c2 < CC; ++c2) acc += (float)a[c2] * (float)b[c2];
        d = acc;
    }
    float mx = d;
    #pragma unroll
    for (int m = 1; m < 32; m <<= 1) mx = fmaxf(mx, __shfl_xor(mx, m, 32));
    float e = (k < KK) ? __expf(d - mx) : 0.f;
    float s = e;
    #pragma unroll
    for (int m = 1; m < 32; m <<= 1) s += __shfl_xor(s, m, 32);
    if (k < KK) simsc[k * PP + p] = e / s;   // layout [KK][P]
}

// ---------------------------------------------------------------------------
// 5b) point-cloud local cosine softmax over knn neighbors. layout [N][KK].
// ---------------------------------------------------------------------------
__global__ __launch_bounds__(32) void pc_local_sim(
    const _Float16* __restrict__ F, const int* __restrict__ nb,
    float* __restrict__ simpc) {
    int i = blockIdx.x;
    int k = threadIdx.x;
    float d = -3.4e38f;
    if (k < KK) {
        int j = nb[i * KK + k];
        const _Float16* a = F + i * CC;
        const _Float16* b = F + j * CC;
        float acc = 0.f;
        for (int c2 = 0; c2 < CC; ++c2) acc += (float)a[c2] * (float)b[c2];
        d = acc;
    }
    float mx = d;
    #pragma unroll
    for (int m = 1; m < 32; m <<= 1) mx = fmaxf(mx, __shfl_xor(mx, m, 32));
    float e = (k < KK) ? __expf(d - mx) : 0.f;
    float s = e;
    #pragma unroll
    for (int m = 1; m < 32; m <<= 1) s += __shfl_xor(s, m, 32);
    if (k < KK) simpc[i * KK + k] = e / s;
}

// ---------------------------------------------------------------------------
// 6) select[k][p] = oob ? NEG : sim_score[k][p] * best_score[edge-clamped nb]
// ---------------------------------------------------------------------------
__global__ void select_kernel(const float* __restrict__ simsc,
                              const unsigned long long* __restrict__ rowbest,
                              float* __restrict__ out) {
    int tid = blockIdx.x * blockDim.x + threadIdx.x;
    if (tid >= KK * PP) return;
    int k = tid / PP, p = tid % PP;
    int h = p / WW, w = p % WW;
    int hn = h + k / 5 - 2, wn = w + k % 5 - 2;
    bool oob = (hn < 0 || hn >= HH || wn < 0 || wn >= WW);
    int hc = iclamp(hn, 0, HH - 1), wc = iclamp(wn, 0, WW - 1);
    float conf = key_score(rowbest[hc * WW + wc]);
    out[tid] = oob ? NEGV : simsc[tid] * conf;
}

// ---------------------------------------------------------------------------
// top_k(0/1 row, 125): sorted unique indices (mask=1), then smallest
// missing indices (mask=0). Exact lowest-index tie-break reproduction.
// ---------------------------------------------------------------------------
__device__ void unique_reduce125(const int* __restrict__ v,
                                 float* __restrict__ mask,
                                 int* __restrict__ idx) {
    int u[SPOT * KK];
    int U = 0;
    for (int j = 0; j < SPOT * KK; ++j) {
        int x = v[j];
        int pos = 0; bool dup = false;
        for (pos = 0; pos < U; ++pos) {
            if (u[pos] == x) { dup = true; break; }
            if (u[pos] > x) break;
        }
        if (dup) continue;
        for (int t = U; t > pos; --t) u[t] = u[t - 1];
        u[pos] = x; ++U;
    }
    for (int j = 0; j < U; ++j) { mask[j] = 1.0f; idx[j] = u[j]; }
    int c = 0, ptr = 0;
    for (int j = U; j < SPOT * KK; ++j) {
        for (;;) {
            if (ptr < U && u[ptr] == c) { ++ptr; ++c; }
            else if (ptr < U && u[ptr] < c) { ++ptr; }
            else break;
        }
        mask[j] = 0.0f; idx[j] = c; ++c;
    }
}

// pick top-(SPOT-1) from 25 values, lowest index wins ties; slots 1..4
__device__ void top4(float* __restrict__ sel, int* __restrict__ tk) {
    for (int s = 1; s < SPOT; ++s) {
        float best = -3.4e38f; int bk = 0;
        for (int k = 0; k < KK; ++k)
            if (sel[k] > best) { best = sel[k]; bk = k; }
        tk[s] = bk;
        sel[bk] = -3.4e38f;
    }
}

// ---------------------------------------------------------------------------
// 7a) image seeding + spotting + reduce: one thread per pixel.
// ---------------------------------------------------------------------------
__global__ void im_seed_spot(const float* __restrict__ select_,
                             const unsigned long long* __restrict__ rowbest,
                             const int* __restrict__ nb,
                             float* __restrict__ mask_out,
                             int* __restrict__ idx_out) {
    int p = blockIdx.x * blockDim.x + threadIdx.x;
    if (p >= PP) return;
    int h = p / WW, w = p % WW;
    float sel[KK];
    for (int k = 0; k < KK; ++k)
        sel[k] = (k == KK / 2) ? NEGV : select_[k * PP + p];
    int tki[SPOT]; tki[0] = KK / 2;
    top4(sel, tki);
    int spot[SPOT * KK];
    for (int j = 0; j < SPOT; ++j) {
        int k = tki[j];
        int hn = h + k / 5 - 2, wn = w + k % 5 - 2;
        int ip = (hn < 0 || hn >= HH || wn < 0 || wn >= WW) ? -1 : hn * WW + wn;
        int np = iclamp(ip, 0, PP - 1);
        int seed = key_idx(rowbest[np]);            // best pc index for pixel
        for (int t = 0; t < KK; ++t) spot[j * KK + t] = nb[seed * KK + t];
    }
    unique_reduce125(spot, mask_out + p * SPOT * KK, idx_out + p * SPOT * KK);
}

// ---------------------------------------------------------------------------
// 7b) pc seeding + spotting + reduce: one thread per point.
// ---------------------------------------------------------------------------
__global__ void pc_seed_spot(const float* __restrict__ simpc,
                             const int* __restrict__ nb,
                             const unsigned long long* __restrict__ colbest,
                             float* __restrict__ mask_out,
                             int* __restrict__ idx_out) {
    int i = blockIdx.x * blockDim.x + threadIdx.x;
    if (i >= NN) return;
    float sel[KK];
    for (int k = 0; k < KK; ++k) {
        int nk = nb[i * KK + k];
        float bscore = key_score(colbest[nk]);
        sel[k] = (k == 0) ? NEGV : simpc[i * KK + k] * bscore;
    }
    int tkp[SPOT]; tkp[0] = 0;
    top4(sel, tkp);
    int spot[SPOT * KK];
    for (int j = 0; j < SPOT; ++j) {
        int nk = nb[i * KK + tkp[j]];
        int q = key_idx(colbest[nk]);               // best pixel index
        int r0 = q / WW, c0 = q % WW;
        for (int t = 0; t < KK; ++t) {
            int r = iclamp(r0 + t / 5 - 2, 0, HH - 1);
            int c = iclamp(c0 + t % 5 - 2, 0, WW - 1);
            spot[j * KK + t] = r * WW + c;
        }
    }
    unique_reduce125(spot, mask_out + i * SPOT * KK, idx_out + i * SPOT * KK);
}

// ---------------------------------------------------------------------------
extern "C" void kernel_launch(void* const* d_in, const int* in_sizes, int n_in,
                              void* d_out, int out_size, void* d_ws, size_t ws_size,
                              hipStream_t stream) {
    (void)in_sizes; (void)n_in; (void)out_size; (void)ws_size;
    const float* imf = (const float*)d_in[0];   // (1,P,C)
    const float* pcf = (const float*)d_in[1];   // (1,N,C)
    const float* pts = (const float*)d_in[2];   // (1,N,3)

    // workspace carve-up (all 256B aligned)
    char* ws = (char*)d_ws;
    _Float16* imf16 = (_Float16*)(ws);                          // P*C*2
    _Float16* pcf16 = (_Float16*)(ws + 1310720);                // N*C*2
    unsigned long long* rowbest = (unsigned long long*)(ws + 2621440);  // P*8
    unsigned long long* colbest = (unsigned long long*)(ws + 2662400);  // N*8
    float* simsc = (float*)(ws + 2703360);                      // KK*P*4
    float* simpc = (float*)(ws + 3215360);                      // N*KK*4
    int*   nb    = (int*)(ws + 3727360);                        // N*KK*4
    // total ~4.24 MB

    // output carve-up (flat, return order)
    float* out_select  = (float*)d_out;                         // 25*P
    float* out_mask    = (float*)d_out + KK * PP;               // P*125
    int*   out_idx     = (int*)((float*)d_out + KK * PP + PP * SPOT * KK);
    int*   out_idx_pc  = (int*)((float*)d_out + KK * PP + 2 * PP * SPOT * KK);
    float* out_mask_pc = (float*)d_out + KK * PP + 3 * PP * SPOT * KK;

    // 1) normalize
    l2norm_rows<<<PP, 128, 0, stream>>>(imf, imf16);
    l2norm_rows<<<NN, 128, 0, stream>>>(pcf, pcf16);
    // 2) init argmax accumulators
    init_best<<<(PP + 255) / 256, 256, 0, stream>>>(rowbest, colbest);
    // 3) WMMA GEMM + fused row/col argmax (80x80 64x64 blocks, 4 waves/block)
    gemm_argmax<<<(NN / 64) * (NN / 64) / 4, 128, 0, stream>>>(
        imf16, pcf16, rowbest, colbest);
    // 4) knn
    knn25<<<NN / 256, 256, 0, stream>>>(pts, nb);
    // 5) local softmaxes
    im_local_sim<<<PP, 32, 0, stream>>>(imf16, simsc);
    pc_local_sim<<<NN, 32, 0, stream>>>(pcf16, nb, simpc);
    // 6) select
    select_kernel<<<(KK * PP + 255) / 256, 256, 0, stream>>>(
        simsc, rowbest, out_select);
    // 7) seeding + spotting
    im_seed_spot<<<(PP + 255) / 256, 256, 0, stream>>>(
        out_select, rowbest, nb, out_mask, out_idx);
    pc_seed_spot<<<(NN + 255) / 256, 256, 0, stream>>>(
        simpc, nb, colbest, out_mask_pc, out_idx_pc);
}
